// VLSTMModel_11776800325719
// MI455X (gfx1250) — compile-verified, matrix operand order checked by reading, not run on hardware
//
#include <hip/hip_runtime.h>

// Problem constants (from reference)
#define T_STEPS 19
#define NBATCH  65536
#define RNNDIM  256
#define EMBDIM  128
#define OUTDIM  5
#define MTILE   64      // batch rows per workgroup
#define THREADS 256     // 8 waves (wave32)

typedef __attribute__((ext_vector_type(16))) _Float16 v16h;
typedef __attribute__((ext_vector_type(8)))  float    v8f;

// WMMA 16-bit A/B fragment layout (CDNA5 ISA 7.12.2):
// lanes 0-15 hold K {0..7,16..23}, lanes 16-31 hold K {8..15,24..31}.
// half-index within a 512-half (32 lane x 16 half) fragment for (m-in-tile, k-in-tile)
__device__ __forceinline__ int aPos(int m16, int kk) {
    int hi = (kk >> 3) & 1;                 // which half-wave holds this k
    int s  = (kk & 7) | ((kk >> 4) << 3);   // slot within the 16 halves
    return (m16 + (hi << 4)) * 16 + s;
}

// ---------------------------------------------------------------------------
// One-time weight packing: B = [W_ih^T ; W_hh^T] (384 x 1024) as f16 in
// per-fragment layout, columns regrouped into 4 blocks of 256:
// block b covers hidden units [b*64, b*64+64) with columns [i|f|g|o] (64 each).
// Bpack[b][kt(12)][tile(16)][lane(32)][slot(16)]  +  packed gate bias [4][256].
// ---------------------------------------------------------------------------
__global__ void pack_kernel(const float* __restrict__ Wih, const float* __restrict__ Whh,
                            const float* __restrict__ bih, const float* __restrict__ bhh,
                            _Float16* __restrict__ Bpack, float* __restrict__ biasPack) {
    int idx = blockIdx.x * 256 + threadIdx.x;
    const int TOTAL = 4 * 12 * 16 * 512;
    if (idx < TOTAL) {
        int s    = idx & 15;
        int r1   = idx >> 4;
        int lane = r1 & 31;
        int r2   = r1 >> 5;
        int tile = r2 & 15;
        int r3   = r2 >> 4;
        int kt   = r3 % 12;
        int b    = r3 / 12;

        int col  = tile * 16 + (lane & 15);    // column within 256-col block
        int gate = col >> 6;                   // 0..3 = i,f,g,o
        int ul   = col & 63;
        int row  = gate * 256 + b * 64 + ul;   // original row of W_ih / W_hh

        int kk = (s & 7) + ((s >> 3) << 4) + ((lane >> 4) << 3);
        int k  = kt * 32 + kk;                 // 0..383
        float v = (k < 128) ? Wih[row * 128 + k] : Whh[row * 256 + (k - 128)];
        Bpack[idx] = (_Float16)v;
    }
    if (idx < 1024) {
        int b = idx >> 8, col = idx & 255;
        int gate = col >> 6, ul = col & 63;
        int row = gate * 256 + b * 64 + ul;
        biasPack[idx] = bih[row] + bhh[row];
    }
}

// ---------------------------------------------------------------------------
// Persistent batch-tile LSTM: h (f16, WMMA fragment layout, double buffered)
// and c (f32) live in LDS for all 19 steps. Gates via v_wmma_f32_16x16x32_f16,
// weights streamed from L2 in fragment order.
// ---------------------------------------------------------------------------
__global__ __launch_bounds__(THREADS, 1)
void lstm_kernel(const float* __restrict__ x, const float* __restrict__ h0,
                 const float* __restrict__ c0, const int* __restrict__ mask,
                 const float* __restrict__ Wemb, const float* __restrict__ bemb,
                 const float* __restrict__ Wout, const float* __restrict__ bout,
                 const _Float16* __restrict__ Bpack, const float* __restrict__ biasPack,
                 float* __restrict__ out) {
    __shared__ _Float16 sE[4 * 4 * 512];        // e: 4 row tiles x 4 k tiles (K 0..127)
    __shared__ _Float16 sH[2][4 * 8 * 512];     // h: double buffer, 4 row x 8 k tiles
    __shared__ float    sC[MTILE * RNNDIM];     // cell state f32
    __shared__ float    sG[MTILE * 256];        // gate staging for one column block
    __shared__ float    sBias[1024];
    __shared__ float    sWemb[2 * EMBDIM];
    __shared__ float    sBemb[EMBDIM];
    __shared__ float    sWout[OUTDIM * RNNDIM];
    __shared__ float    sBout[8];
    __shared__ float    sX[MTILE * 2];
    __shared__ float    sM[MTILE];

    const int tid  = threadIdx.x;
    const int w    = tid >> 5;
    const int lane = tid & 31;
    const int rowBase = blockIdx.x * MTILE;
    const size_t OFF_H = (size_t)T_STEPS * NBATCH * OUTDIM;
    const size_t OFF_C = OFF_H + (size_t)NBATCH * RNNDIM;

    // ---- one-time staging ----
    for (int i = tid; i < EMBDIM; i += THREADS) {
        sWemb[i] = Wemb[i];
        sWemb[EMBDIM + i] = Wemb[EMBDIM + i];
        sBemb[i] = bemb[i];
    }
    for (int i = tid; i < 1024; i += THREADS) sBias[i] = biasPack[i];
    for (int i = tid; i < OUTDIM * RNNDIM; i += THREADS) sWout[i] = Wout[i];
    if (tid < OUTDIM) sBout[tid] = bout[tid];
    for (int p = tid; p < MTILE * RNNDIM; p += THREADS) {
        int m = p >> 8, u = p & 255;
        int row = rowBase + m;
        sC[p] = c0[(size_t)row * RNNDIM + u];
        float hv = h0[(size_t)row * RNNDIM + u];
        sH[0][((m >> 4) * 8 + (u >> 5)) * 512 + aPos(m & 15, u & 31)] = (_Float16)hv;
    }
    __syncthreads();

    int cur = 0;
    for (int t = 0; t < T_STEPS; ++t) {
        // ---- stage x, mask; compute e = relu(x @ Wemb + bemb) into fragment layout
        for (int i = tid; i < MTILE; i += THREADS) {
            int row = rowBase + i;
            sX[i * 2 + 0] = x[((size_t)t * NBATCH + row) * 2 + 0];
            sX[i * 2 + 1] = x[((size_t)t * NBATCH + row) * 2 + 1];
            sM[i] = (mask[(size_t)t * NBATCH + row] == 1) ? 1.0f : 0.0f;
        }
        __syncthreads();
        for (int p = tid; p < MTILE * EMBDIM; p += THREADS) {
            int m = p >> 7, j = p & 127;
            float e = sX[m * 2] * sWemb[j] + sX[m * 2 + 1] * sWemb[EMBDIM + j] + sBemb[j];
            e = e > 0.0f ? e : 0.0f;
            sE[((m >> 4) * 4 + (j >> 5)) * 512 + aPos(m & 15, j & 31)] = (_Float16)e;
        }
        __syncthreads();

        const _Float16* hb = sH[cur];
        _Float16*       hn = sH[cur ^ 1];

        // ---- 4 column blocks of 256 gate columns: GEMM + fused cell update
        for (int b = 0; b < 4; ++b) {
            v8f acc[4][2] = {};
            for (int kt = 0; kt < 12; ++kt) {
                const _Float16* bbase =
                    Bpack + ((((size_t)b * 12 + kt) * 16 + w * 2) * 512) + lane * 16;
                if (kt + 1 < 12) __builtin_prefetch(bbase + 16 * 512, 0, 0);
                v16h bf0 = *(const v16h*)(bbase);
                v16h bf1 = *(const v16h*)(bbase + 512);
#pragma unroll
                for (int rt = 0; rt < 4; ++rt) {
                    const _Float16* ap = (kt < 4)
                        ? &sE[(rt * 4 + kt) * 512 + lane * 16]
                        : &hb[(rt * 8 + (kt - 4)) * 512 + lane * 16];
                    v16h af = *(const v16h*)ap;
                    acc[rt][0] = __builtin_amdgcn_wmma_f32_16x16x32_f16(
                        false, af, false, bf0, (short)0, acc[rt][0], false, false);
                    acc[rt][1] = __builtin_amdgcn_wmma_f32_16x16x32_f16(
                        false, af, false, bf1, (short)0, acc[rt][1], false, false);
                }
            }
            // scatter accumulators to gate staging (C layout: m = r + 8*(lane>=16))
#pragma unroll
            for (int rt = 0; rt < 4; ++rt) {
#pragma unroll
                for (int ct = 0; ct < 2; ++ct) {
                    int col = (w * 2 + ct) * 16 + (lane & 15);
                    int mB  = rt * 16 + ((lane >> 4) << 3);
#pragma unroll
                    for (int r = 0; r < 8; ++r)
                        sG[(mB + r) * 256 + col] = acc[rt][ct][r];
                }
            }
            __syncthreads();

            // fused LSTM cell update for hidden units [b*64, b*64+64)
            for (int p = tid; p < MTILE * 64; p += THREADS) {
                int m = p >> 6, ul = p & 63;
                float iv = sG[m * 256 + ul]       + sBias[b * 256 + ul];
                float fv = sG[m * 256 + 64 + ul]  + sBias[b * 256 + 64 + ul];
                float gv = sG[m * 256 + 128 + ul] + sBias[b * 256 + 128 + ul];
                float ov = sG[m * 256 + 192 + ul] + sBias[b * 256 + 192 + ul];
                float ig = 1.0f / (1.0f + __expf(-iv));
                float fg = 1.0f / (1.0f + __expf(-fv));
                float gg = tanhf(gv);
                float og = 1.0f / (1.0f + __expf(-ov));
                int u = b * 64 + ul;
                float cOld = sC[m * 256 + u];
                float cNew = fg * cOld + ig * gg;
                float hNew = og * tanhf(cNew);
                float mf = sM[m];
                int pos = ((m >> 4) * 8 + (u >> 5)) * 512 + aPos(m & 15, u & 31);
                float hOld = (float)hb[pos];
                sC[m * 256 + u] = cOld + mf * (cNew - cOld);
                hn[pos] = (_Float16)(hOld + mf * (hNew - hOld));
            }
            __syncthreads();
        }

        // ---- output projection: out = mask * (h_t @ Wout^T + bout), OUT=5
        for (int task = tid; task < MTILE * OUTDIM; task += THREADS) {
            int m = task / OUTDIM, o = task - m * OUTDIM;
            float s = sBout[o];
            for (int u = 0; u < RNNDIM; ++u) {
                int pos = ((m >> 4) * 8 + (u >> 5)) * 512 + aPos(m & 15, u & 31);
                s += (float)hn[pos] * sWout[o * RNNDIM + u];
            }
            int row = rowBase + m;
            out[(size_t)t * NBATCH * OUTDIM + (size_t)row * OUTDIM + o] = sM[m] * s;
        }
        __syncthreads();
        cur ^= 1;
    }

    // ---- final h, c
    for (int p = tid; p < MTILE * RNNDIM; p += THREADS) {
        int m = p >> 8, u = p & 255;
        int row = rowBase + m;
        out[OFF_C + (size_t)row * RNNDIM + u] = sC[p];
        int pos = ((m >> 4) * 8 + (u >> 5)) * 512 + aPos(m & 15, u & 31);
        out[OFF_H + (size_t)row * RNNDIM + u] = (float)sH[cur][pos];
    }
}

extern "C" void kernel_launch(void* const* d_in, const int* in_sizes, int n_in,
                              void* d_out, int out_size, void* d_ws, size_t ws_size,
                              hipStream_t stream) {
    const float* x    = (const float*)d_in[0];
    const float* h0   = (const float*)d_in[1];
    const float* c0   = (const float*)d_in[2];
    const int*   mask = (const int*)d_in[3];
    const float* Wemb = (const float*)d_in[4];
    const float* bemb = (const float*)d_in[5];
    const float* Wih  = (const float*)d_in[6];
    const float* bih  = (const float*)d_in[7];
    const float* Whh  = (const float*)d_in[8];
    const float* bhh  = (const float*)d_in[9];
    const float* Wout = (const float*)d_in[10];
    const float* bout = (const float*)d_in[11];

    _Float16* Bpack = (_Float16*)d_ws;
    float* biasPack = (float*)((char*)d_ws + (size_t)4 * 12 * 16 * 512 * sizeof(_Float16));

    pack_kernel<<<(4 * 12 * 16 * 512 + 255) / 256, 256, 0, stream>>>(
        Wih, Whh, bih, bhh, Bpack, biasPack);

    lstm_kernel<<<NBATCH / MTILE, THREADS, 0, stream>>>(
        x, h0, c0, mask, Wemb, bemb, Wout, bout, Bpack, biasPack, (float*)d_out);
}
